// SAM_58317065945994
// MI455X (gfx1250) — matrix operand order repaired and checked by reading, not compiled, not run on hardware
//
#include <hip/hip_runtime.h>

#define NPIX 4096
#define CIN  32
#define CQK  64
#define BATCH 8

typedef __attribute__((ext_vector_type(16))) _Float16 v16h;
typedef __attribute__((ext_vector_type(8)))  _Float16 v8h;
typedef __attribute__((ext_vector_type(8)))  float    v8f;
typedef __attribute__((ext_vector_type(4)))  float    f4v;

// Assemble a 16-half WMMA operand from two 16-byte chunks.
__device__ __forceinline__ v16h load_frag(const _Float16* p0, const _Float16* p1) {
    v8h a = *(const v8h*)p0;
    v8h b = *(const v8h*)p1;
    return __builtin_shufflevector(a, b, 0,1,2,3,4,5,6,7,8,9,10,11,12,13,14,15);
}

__device__ __forceinline__ v8f wmma_f16(v16h a, v16h b, v8f c) {
    return __builtin_amdgcn_wmma_f32_16x16x32_f16(false, a, false, b, (short)0, c, false, false);
}

// Single-instruction 16-lane butterfly steps: VALU op with DPP16 ROW_XMASK on src0.
// (modifier string matches what the compiler itself emits for gfx1250)
#define DPP_MAX_STEP(x, MSTR) do { float _t;                                              \
    asm("v_max_num_f32_dpp %0, %1, %1 row_xmask:" MSTR                                    \
        " row_mask:0xf bank_mask:0xf bound_ctrl:1" : "=v"(_t) : "v"(x)); x = _t; } while (0)
#define DPP_ADD_STEP(x, MSTR) do { float _t;                                              \
    asm("v_add_f32_dpp %0, %1, %1 row_xmask:" MSTR                                        \
        " row_mask:0xf bank_mask:0xf bound_ctrl:1" : "=v"(_t) : "v"(x)); x = _t; } while (0)

// max without canonicalize overhead (v_max_num ignores NaN per IEEE maximumNumber)
__device__ __forceinline__ float fmax_asm(float a, float b) {
    float r; asm("v_max_num_f32 %0, %1, %2" : "=v"(r) : "v"(a), "v"(b)); return r;
}
__device__ __forceinline__ float fmax3_asm(float a, float b, float c) {
    float r; asm("v_max3_num_f32 %0, %1, %2, %3" : "=v"(r) : "v"(a), "v"(b), "v"(c)); return r;
}

// Async-copy one 64-key K/V tile (16 KB) into LDS: 4 x b128 per thread.
// Tracked by ASYNCcnt; completion enforced with s_wait_asynccnt + barrier.
__device__ __forceinline__ void issue_tile_async(
    const _Float16* __restrict__ Kb, const _Float16* __restrict__ Vb, int key0,
    _Float16* kb, _Float16* vb, int tid)
{
    #pragma unroll
    for (int i = 0; i < 2; ++i) {
        int s   = tid + i * 256;          // 0..511 segments of 16 B
        int row = s >> 3;                 // 0..63
        int off = (s & 7) * 8;            // halves within 128 B row
        {   // K tile: 64 d-rows x 64 keys
            unsigned long long g = (unsigned long long)(uintptr_t)(Kb + (size_t)row * NPIX + key0 + off);
            unsigned int       l = (unsigned int)(uintptr_t)(kb + row * 64 + off);
            asm volatile("global_load_async_to_lds_b128 %0, %1, off" :: "v"(l), "v"(g) : "memory");
        }
        {   // V tile: 64 key-rows x 64 dims
            unsigned long long g = (unsigned long long)(uintptr_t)(Vb + (size_t)(key0 + row) * CQK + off);
            unsigned int       l = (unsigned int)(uintptr_t)(vb + row * 64 + off);
            asm volatile("global_load_async_to_lds_b128 %0, %1, off" :: "v"(l), "v"(g) : "memory");
        }
    }
}

__device__ __forceinline__ void wait_async_and_barrier() {
    asm volatile("s_wait_asynccnt 0x0" ::: "memory");
    __syncthreads();
}

// ---------------------------------------------------------------------------
// Phase 1: 1x1-conv projections. x is (B, CIN, N) channel-major.
// Q: (B, N, CQK) f16 ; Kt: (B, CQK, N) f16 (pre-transposed) ; V: (B, N, CQK) f16
// ---------------------------------------------------------------------------
__global__ __launch_bounds__(256) void qkv_kernel(
    const float* __restrict__ x,
    const float* __restrict__ wq, const float* __restrict__ bq,
    const float* __restrict__ wk, const float* __restrict__ bk,
    const float* __restrict__ wv, const float* __restrict__ bv,
    _Float16* __restrict__ Qh, _Float16* __restrict__ Kt, _Float16* __restrict__ Vh)
{
    __shared__ float swq[CQK*CIN], swk[CQK*CIN], swv[CQK*CIN];
    __shared__ float sbq[CQK], sbk[CQK], sbv[CQK];
    for (int i = threadIdx.x; i < CQK*CIN; i += 256) {
        swq[i] = wq[i]; swk[i] = wk[i]; swv[i] = wv[i];
    }
    if (threadIdx.x < CQK) {
        sbq[threadIdx.x] = bq[threadIdx.x];
        sbk[threadIdx.x] = bk[threadIdx.x];
        sbv[threadIdx.x] = bv[threadIdx.x];
    }
    __syncthreads();

    int g = blockIdx.x * 256 + threadIdx.x;   // 0 .. B*N-1
    int b = g >> 12;
    int n = g & (NPIX - 1);

    float xv[CIN];
    const float* xb = x + (size_t)b * CIN * NPIX + n;
    #pragma unroll
    for (int c = 0; c < CIN; ++c) xv[c] = xb[(size_t)c * NPIX];

    size_t qoff = ((size_t)b * NPIX + n) * CQK;
    size_t koff = (size_t)b * CQK * NPIX + n;

    #pragma unroll 4
    for (int o = 0; o < CQK; ++o) {
        float aq = sbq[o], ak = sbk[o], av = sbv[o];
        const float* wqo = &swq[o*CIN];
        const float* wko = &swk[o*CIN];
        const float* wvo = &swv[o*CIN];
        #pragma unroll
        for (int c = 0; c < CIN; ++c) {
            float xc = xv[c];
            aq += xc * wqo[c];
            ak += xc * wko[c];
            av += xc * wvo[c];
        }
        Qh[qoff + o]                 = (_Float16)aq;
        Kt[koff + (size_t)o * NPIX]  = (_Float16)ak;
        Vh[qoff + o]                 = (_Float16)av;
    }
}

// ---------------------------------------------------------------------------
// Phase 2: fused flash attention. One wave = 16 query rows, 8 waves/block.
// 64-key tiles, K/V double-buffered in LDS via async global->LDS copies.
// grid.x = BATCH * (NPIX/128)
// ---------------------------------------------------------------------------
__global__ __launch_bounds__(256) void attn_kernel(
    const _Float16* __restrict__ Qh, const _Float16* __restrict__ Kt,
    const _Float16* __restrict__ Vh, float* __restrict__ Ob)
{
    __shared__ alignas(16) _Float16 kbuf[2][CQK * 64];   // 2 x 8 KB, [d 0..63][key 0..63]
    __shared__ alignas(16) _Float16 vbuf[2][64 * CQK];   // 2 x 8 KB, [key 0..63][d 0..63]
    __shared__ alignas(16) _Float16 pbuf[8 * 16 * 64];   // per-wave P staging, 16 KB

    const int tid  = threadIdx.x;
    const int wave = tid >> 5;
    const int lane = tid & 31;
    const int l16  = lane & 15;
    const bool hi  = lane >= 16;

    const int b     = blockIdx.x >> 5;       // 32 row-tiles of 128 per batch
    const int tile  = blockIdx.x & 31;
    const int qrow0 = tile * 128 + wave * 16;

    const _Float16* Qb = Qh + ((size_t)b * NPIX + qrow0) * CQK;
    const _Float16* Kb = Kt + (size_t)b * CQK * NPIX;
    const _Float16* Vb = Vh + (size_t)b * NPIX * CQK;

    // A-fragment layout (16-bit, 16x32): lane m -> K {0..7,16..23}, lane m+16 -> {8..15,24..31}
    const int c0 = hi ? 8 : 0;
    const _Float16* qrow = Qb + (size_t)l16 * CQK;
    v16h aq0 = load_frag(qrow + c0,      qrow + c0 + 16);      // d 0..31
    v16h aq1 = load_frag(qrow + 32 + c0, qrow + 32 + c0 + 16); // d 32..63

    v8f o[4] = {};
    float m[8], l[8];
    #pragma unroll
    for (int r = 0; r < 8; ++r) { m[r] = -1e30f; l[r] = 0.0f; }

    _Float16* mybuf = pbuf + wave * (16 * 64);
    const int drow = (hi ? 16 : 0) + l16;   // B-fragment K-row within 32-chunk
    const int vrow = (hi ? 16 : 0) + l16;

    // prologue: stage tile 0
    issue_tile_async(Kb, Vb, 0, kbuf[0], vbuf[0], tid);
    wait_async_and_barrier();

    int parity = 0;
    for (int key0 = 0; key0 < NPIX; key0 += 64, parity ^= 1) {
        const _Float16* kb = kbuf[parity];
        const _Float16* vb = vbuf[parity];

        // software pipeline: kick off next tile while computing this one
        if (key0 + 64 < NPIX)
            issue_tile_async(Kb, Vb, key0 + 64, kbuf[parity ^ 1], vbuf[parity ^ 1], tid);

        // S = Q * K^T : 4 key-groups of 16, accumulate over d = 2 x 32
        v8f s[4];
        #pragma unroll
        for (int kg = 0; kg < 4; ++kg) {
            v16h b0, b1;
            #pragma unroll
            for (int e = 0; e < 16; ++e) {
                int d = (e < 8 ? e : e + 8) + c0;
                b0[e] = kb[(size_t)d        * 64 + kg * 16 + l16];  // d 0..31
                b1[e] = kb[(size_t)(32 + d) * 64 + kg * 16 + l16];  // d 32..63
            }
            v8f z = {};
            z     = wmma_f16(aq0, b0, z);
            s[kg] = wmma_f16(aq1, b1, z);
        }

        // --- online softmax: rows in VGPR index, cols across 16 lanes ---
        float mt[8];
        #pragma unroll
        for (int r = 0; r < 8; ++r)
            mt[r] = fmax_asm(fmax3_asm(s[0][r], s[1][r], s[2][r]), s[3][r]);
        #pragma unroll
        for (int r = 0; r < 8; ++r) DPP_MAX_STEP(mt[r], "1");
        #pragma unroll
        for (int r = 0; r < 8; ++r) DPP_MAX_STEP(mt[r], "2");
        #pragma unroll
        for (int r = 0; r < 8; ++r) DPP_MAX_STEP(mt[r], "4");
        #pragma unroll
        for (int r = 0; r < 8; ++r) DPP_MAX_STEP(mt[r], "8");

        float scale[8], lt[8];
        #pragma unroll
        for (int r = 0; r < 8; ++r) {
            float mn = fmax_asm(m[r], mt[r]);
            scale[r] = __expf(m[r] - mn);
            m[r] = mn;
            float p0 = __expf(s[0][r] - mn);
            float p1 = __expf(s[1][r] - mn);
            float p2 = __expf(s[2][r] - mn);
            float p3 = __expf(s[3][r] - mn);
            s[0][r] = p0; s[1][r] = p1; s[2][r] = p2; s[3][r] = p3;
            lt[r] = (p0 + p1) + (p2 + p3);
        }
        #pragma unroll
        for (int r = 0; r < 8; ++r) DPP_ADD_STEP(lt[r], "1");
        #pragma unroll
        for (int r = 0; r < 8; ++r) DPP_ADD_STEP(lt[r], "2");
        #pragma unroll
        for (int r = 0; r < 8; ++r) DPP_ADD_STEP(lt[r], "4");
        #pragma unroll
        for (int r = 0; r < 8; ++r) DPP_ADD_STEP(lt[r], "8");

        #pragma unroll
        for (int r = 0; r < 8; ++r) {
            l[r] = l[r] * scale[r] + lt[r];
            o[0][r] *= scale[r]; o[1][r] *= scale[r];
            o[2][r] *= scale[r]; o[3][r] *= scale[r];
        }

        // --- transpose P: D layout -> A layout via wave-private LDS ---
        #pragma unroll
        for (int r = 0; r < 8; ++r) {
            int prow = hi ? (r + 8) : r;
            _Float16* pr = mybuf + prow * 64 + l16;
            pr[0]  = (_Float16)s[0][r];
            pr[16] = (_Float16)s[1][r];
            pr[32] = (_Float16)s[2][r];
            pr[48] = (_Float16)s[3][r];
        }
        // same-wave DS ops are in-order: no barrier needed (wave-private region)
        const _Float16* pp = mybuf + l16 * 64;
        v16h ap0 = load_frag(pp + c0,      pp + c0 + 16);      // keys 0..31
        v16h ap1 = load_frag(pp + 32 + c0, pp + 32 + c0 + 16); // keys 32..63

        // O += P * V : 4 output-dim chunks, accumulate over 2 x 32 keys
        #pragma unroll
        for (int c = 0; c < 4; ++c) {
            v16h bvA, bvB;
            #pragma unroll
            for (int e = 0; e < 16; ++e) {
                int kk = (e < 8 ? e : e + 8) + c0;
                bvA[e] = vb[(size_t)kk        * 64 + c*16 + l16];    // key rows 0..31
                bvB[e] = vb[(size_t)(32 + kk) * 64 + c*16 + l16];    // key rows 32..63
            }
            o[c] = wmma_f16(ap0, bvA, o[c]);
            o[c] = wmma_f16(ap1, bvB, o[c]);
        }

        // next tile landed + everyone done reading current buffers
        wait_async_and_barrier();
    }

    // epilogue: normalize and write O (B, N, CQK) f32
    float* ob = Ob + ((size_t)b * NPIX + qrow0) * CQK;
    #pragma unroll
    for (int r = 0; r < 8; ++r) {
        float inv = 1.0f / l[r];
        int prow = hi ? (r + 8) : r;
        float* row = ob + (size_t)prow * CQK + l16;
        row[0]  = o[0][r] * inv;
        row[16] = o[1][r] * inv;
        row[32] = o[2][r] * inv;
        row[48] = o[3][r] * inv;
    }
}

// ---------------------------------------------------------------------------
// Phase 3: output projection + bias + residual. out (B, CIN, N) f32.
// ---------------------------------------------------------------------------
__global__ __launch_bounds__(256) void proj_kernel(
    const float* __restrict__ Ob, const float* __restrict__ wo,
    const float* __restrict__ bo, const float* __restrict__ x,
    float* __restrict__ out)
{
    __shared__ float swo[CIN*CQK];
    __shared__ float sbo[CIN];
    for (int i = threadIdx.x; i < CIN*CQK; i += 256) swo[i] = wo[i];
    if (threadIdx.x < CIN) sbo[threadIdx.x] = bo[threadIdx.x];
    __syncthreads();

    int g = blockIdx.x * 256 + threadIdx.x;
    int b = g >> 12;
    int n = g & (NPIX - 1);

    float ov[CQK];
    const float* orow = Ob + ((size_t)b * NPIX + n) * CQK;
    #pragma unroll
    for (int c = 0; c < CQK; c += 4) {
        f4v t = *(const f4v*)(orow + c);
        ov[c] = t.x; ov[c+1] = t.y; ov[c+2] = t.z; ov[c+3] = t.w;
    }

    #pragma unroll 4
    for (int o = 0; o < CIN; ++o) {
        float acc = sbo[o];
        const float* w = &swo[o*CQK];
        #pragma unroll
        for (int c = 0; c < CQK; ++c) acc += ov[c] * w[c];
        size_t idx = ((size_t)b * CIN + o) * NPIX + n;
        out[idx] = acc + x[idx];
    }
}

// ---------------------------------------------------------------------------
extern "C" void kernel_launch(void* const* d_in, const int* in_sizes, int n_in,
                              void* d_out, int out_size, void* d_ws, size_t ws_size,
                              hipStream_t stream)
{
    (void)in_sizes; (void)n_in; (void)out_size; (void)ws_size;
    const float* x  = (const float*)d_in[0];
    const float* wq = (const float*)d_in[1];
    const float* bq = (const float*)d_in[2];
    const float* wk = (const float*)d_in[3];
    const float* bk = (const float*)d_in[4];
    const float* wv = (const float*)d_in[5];
    const float* bv = (const float*)d_in[6];
    const float* wo = (const float*)d_in[7];
    const float* bo = (const float*)d_in[8];

    const size_t nqkv = (size_t)BATCH * NPIX * CQK;   // 2M elements
    _Float16* Qh = (_Float16*)d_ws;
    _Float16* Kt = Qh + nqkv;
    _Float16* Vh = Kt + nqkv;
    float*    Ob = (float*)(Vh + nqkv);               // 8 MB f32

    qkv_kernel<<<(BATCH*NPIX)/256, 256, 0, stream>>>(x, wq, bq, wk, bk, wv, bv, Qh, Kt, Vh);
    attn_kernel<<<BATCH*(NPIX/128), 256, 0, stream>>>(Qh, Kt, Vh, Ob);
    proj_kernel<<<(BATCH*NPIX)/256, 256, 0, stream>>>(Ob, wo, bo, x, (float*)d_out);
}
